// DeepHam_13666585936224
// MI455X (gfx1250) — matrix operand order repaired and checked by reading, not compiled, not run on hardware
//
#include <hip/hip_runtime.h>
#include <hip/hip_bf16.h>

#define NV 512       // vertices
#define DF 32        // feature dim
#define NE 8192      // edges (without self loops)
#define MT (NE + NV) // total edges incl self loops = 8704
#define RS 33        // padded LDS row stride (floats); gcd(33,64)=1 -> conflict-free

typedef __attribute__((ext_vector_type(16))) __bf16 v16bf;
typedef __attribute__((ext_vector_type(8)))  float  v8f;

// ---------------------------------------------------------------------------
// Kernel 1: deterministic CSR by dst (edge order preserved, self-loop last —
// same per-segment summation order as the reference's concat([edges, loops])).
// Outputs: row_ptr[NV+1], src[MT], dinv[NV].  (norm is refactored away:
// norm[e] = dinv[src]*dinv[dst] is applied as row scalings in the main kernel.)
// ---------------------------------------------------------------------------
__global__ __launch_bounds__(1024) void deepham_build_csr(
    const int* __restrict__ edge_index,  // [2][NE]
    int* __restrict__ row_ptr,           // [NV+1]
    int* __restrict__ src_g,             // [MT]
    float* __restrict__ dinv_g)          // [NV]
{
    __shared__ int   dstS[NE];   // 32 KB
    __shared__ int   counts[NV];
    __shared__ int   rp[NV + 1];

    const int tid = threadIdx.x;
    const int* esrc = edge_index;
    const int* edst = edge_index + NE;

    for (int e = tid; e < NE; e += 1024) dstS[e] = edst[e];
    if (tid < NV) counts[tid] = 1;  // self loop
    __syncthreads();
    for (int e = tid; e < NE; e += 1024) atomicAdd(&counts[dstS[e]], 1);
    __syncthreads();
    if (tid < NV) dinv_g[tid] = rsqrtf((float)counts[tid]);
    if (tid == 0) {
        int acc = 0;
        for (int i = 0; i < NV; ++i) { rp[i] = acc; acc += counts[i]; }
        rp[NV] = acc;
    }
    __syncthreads();
    if (tid <= NV) row_ptr[tid] = rp[tid];
    if (tid < NV) {
        int pos = rp[tid];
        for (int e = 0; e < NE; ++e) {
            if (dstS[e] == tid) src_g[pos++] = esrc[e];
        }
        src_g[pos] = tid;   // self loop appended last (matches concat order)
    }
}

// ---------------------------------------------------------------------------
// Persistent main kernel: one workgroup (32 waves), whole 512-step scan in LDS.
// ---------------------------------------------------------------------------

// Build the 16x32 bf16 A fragment for rows 16w..16w+15 of vbuf.
// lanes 0-15: row m=lane,    K = {0..15}
// lanes 16-31: row m=lane-16, K = {16..31}
__device__ __forceinline__ v16bf load_a_frag(const float* __restrict__ vbuf,
                                             int w, int lane)
{
    const int mrow = (w << 4) + (lane & 15);
    const int kb   = (lane < 16) ? 0 : 16;
    v16bf a;
#pragma unroll
    for (int j = 0; j < 8; ++j) {
        a[j]     = (__bf16)vbuf[mrow * RS + kb + j];
        a[j + 8] = (__bf16)vbuf[mrow * RS + kb + 8 + j];
    }
    return a;
}

// h[16w..16w+15][:] = (v @ W) * (SCALE ? dinv[row] : 1), two bf16 WMMAs/wave.
template <bool SCALE>
__device__ __forceinline__ void gemm_tile(const float* __restrict__ vbuf,
                                          float* __restrict__ hbuf,
                                          const __bf16* __restrict__ Wt, // [n][k] bf16 (transposed)
                                          const float* __restrict__ dinvS,
                                          int w, int lane)
{
    const v16bf a  = load_a_frag(vbuf, w, lane);
    const int koff = (lane < 16) ? 0 : 16;
    const int mbase = (w << 4) + ((lane < 16) ? 0 : 8);
    float sc[8];
    if (SCALE) {
#pragma unroll
        for (int r = 0; r < 8; ++r) sc[r] = dinvS[mbase + r];
    }
#pragma unroll
    for (int ct = 0; ct < 2; ++ct) {
        const int n = (ct << 4) + (lane & 15);
        v16bf b = *(const v16bf*)(Wt + n * DF + koff);
        v8f c = {};
        c = __builtin_amdgcn_wmma_f32_16x16x32_bf16(
                false, a, false, b, (short)0, c, false, false);
#pragma unroll
        for (int r = 0; r < 8; ++r)
            hbuf[(mbase + r) * RS + n] = SCALE ? c[r] * sc[r] : c[r];
    }
}

// v[d][lane] = tanh( dinv[d] * sum_{e in CSR[d]} h'[src[e]][lane] + bias )
// where h' rows are pre-scaled by dinv[src] in the gemm epilogue.
__device__ __forceinline__ void aggregate(const float* __restrict__ hbuf,
                                          float* __restrict__ vbuf,
                                          float biasl,
                                          const int* __restrict__ rp,
                                          const unsigned short* __restrict__ srcS,
                                          const float* __restrict__ dinvS,
                                          int w, int lane)
{
    for (int d = w; d < NV; d += 32) {
        const int e0 = rp[d], e1 = rp[d + 1];
        float a0 = 0.f, a1 = 0.f, a2 = 0.f, a3 = 0.f;
        int e = e0;
        for (; e + 4 <= e1; e += 4) {
            const int s0 = srcS[e], s1 = srcS[e + 1];
            const int s2 = srcS[e + 2], s3 = srcS[e + 3];
            a0 += hbuf[s0 * RS + lane];
            a1 += hbuf[s1 * RS + lane];
            a2 += hbuf[s2 * RS + lane];
            a3 += hbuf[s3 * RS + lane];
        }
        for (; e < e1; ++e) a0 += hbuf[srcS[e] * RS + lane];
        const float acc = (a0 + a1) + (a2 + a3);
        vbuf[d * RS + lane] = tanhf(fmaf(acc, dinvS[d], biasl));
    }
}

__global__ __launch_bounds__(1024, 1) void deepham_main(
    const float* __restrict__ vertices,
    const float* __restrict__ W1, const float* __restrict__ b1,
    const float* __restrict__ W2, const float* __restrict__ b2,
    const float* __restrict__ W3, const float* __restrict__ b3,
    const float* __restrict__ Wm1, const float* __restrict__ bm1,
    const float* __restrict__ Wm2, const float* __restrict__ bm2,
    const int* __restrict__ row_ptr_g,
    const int* __restrict__ src_g,
    const float* __restrict__ dinv_g,
    float* __restrict__ out)
{
    __shared__ float v[NV * RS];                   // 67584 B
    __shared__ float h[NV * RS];                   // 67584 B
    __shared__ alignas(64) __bf16 Wt[3][DF * DF];  // 6144 B, [layer][n][k]
    __shared__ int   rp[NV + 1];                   // 2052 B
    __shared__ unsigned short srcS[MT];            // 17408 B
    __shared__ float dinvS[NV];                    // 2048 B
    __shared__ int   chosen;                       // total ~159 KB

    float* probs = h;   // h[0..511] is dead during the MLP/argmax phase

    const int tid  = threadIdx.x;
    const int lane = tid & 31;
    const int w    = tid >> 5;

    // ---- hoisted per-thread constants (loop-invariant) ----
    const float biasr0 = b1[lane], biasr1 = b2[lane], biasr2 = b3[lane];
    const int   nlo = lane & 15;
    const float bm1_0 = bm1[nlo],      bm1_1 = bm1[nlo + 16];
    const float wm2_0 = Wm2[nlo],      wm2_1 = Wm2[nlo + 16];  // Wm2 is (32,1)
    const float bm2v  = bm2[0];

    // Hoisted B fragments for the MLP gemm (Wm1), built straight from global.
    v16bf bmlp[2];
    {
        const int koff = (lane < 16) ? 0 : 16;
#pragma unroll
        for (int ct = 0; ct < 2; ++ct) {
            const int n = (ct << 4) + nlo;
#pragma unroll
            for (int j = 0; j < 16; ++j)
                bmlp[ct][j] = (__bf16)Wm1[(koff + j) * DF + n];
        }
    }

    // ---- one-time staging into LDS ----
    for (int i = tid; i < NV * DF; i += 1024) {
        const int r = i >> 5, c = i & 31;
        v[r * RS + c] = vertices[i];
    }
    for (int i = tid; i < DF * DF; i += 1024) {
        const int k = i >> 5, n = i & 31;      // W is [k][n] row-major
        Wt[0][n * DF + k] = (__bf16)W1[i];
        Wt[1][n * DF + k] = (__bf16)W2[i];
        Wt[2][n * DF + k] = (__bf16)W3[i];
    }
    for (int i = tid; i < NV + 1; i += 1024) rp[i] = row_ptr_g[i];
    for (int i = tid; i < MT; i += 1024) srcS[i] = (unsigned short)src_g[i];
    if (tid < NV) dinvS[tid] = dinv_g[tid];
    __syncthreads();

    // ---- 512 sequential steps, all in LDS ----
    for (int step = 0; step < NV; ++step) {
        gemm_tile<true>(v, h, Wt[0], dinvS, w, lane);              __syncthreads();
        aggregate(h, v, biasr0, rp, srcS, dinvS, w, lane);         __syncthreads();
        gemm_tile<true>(v, h, Wt[1], dinvS, w, lane);              __syncthreads();
        aggregate(h, v, biasr1, rp, srcS, dinvS, w, lane);         __syncthreads();
        gemm_tile<true>(v, h, Wt[2], dinvS, w, lane);              __syncthreads();
        aggregate(h, v, biasr2, rp, srcS, dinvS, w, lane);         __syncthreads();

        // ---- MLP head fused into the 4th WMMA's registers ----
        {
            const v16bf a = load_a_frag(v, w, lane);
            v8f c0 = {}, c1 = {};
            c0 = __builtin_amdgcn_wmma_f32_16x16x32_bf16(
                     false, a, false, bmlp[0], (short)0, c0, false, false);
            c1 = __builtin_amdgcn_wmma_f32_16x16x32_bf16(
                     false, a, false, bmlp[1], (short)0, c1, false, false);
            // probs[row] = sum_n relu(t[row][n]+bm1[n])*wm2[n] + bm2
            float psum[8];
#pragma unroll
            for (int r = 0; r < 8; ++r) {
                float p = fmaxf(c0[r] + bm1_0, 0.f) * wm2_0
                        + fmaxf(c1[r] + bm1_1, 0.f) * wm2_1;
#pragma unroll
                for (int off = 8; off >= 1; off >>= 1)
                    p += __shfl_xor(p, off, 32);   // reduce within each 16-lane half
                psum[r] = p;
            }
            if (nlo == 0) {
                const int rbase = (w << 4) + ((lane < 16) ? 0 : 8);
#pragma unroll
                for (int r = 0; r < 8; ++r)
                    probs[rbase + r] = psum[r] + bm2v;
            }
        }
        __syncthreads();

        // ---- argmax with first-index tie-break (wave 0, wave32 shuffle) ----
        if (w == 0) {
            float best = -__builtin_inff();
            int   bidx = 0;
            for (int i = lane; i < NV; i += 32) {
                const float p = probs[i];
                if (p > best) { best = p; bidx = i; }
            }
#pragma unroll
            for (int off = 16; off > 0; off >>= 1) {
                const float ob = __shfl_down(best, off, 32);
                const int   oi = __shfl_down(bidx, off, 32);
                if (ob > best || (ob == best && oi < bidx)) { best = ob; bidx = oi; }
            }
            if (lane == 0) chosen = bidx;
        }
        __syncthreads();

        if (tid < DF) out[step * DF + tid] = v[chosen * RS + tid];
        // argmax's probs reads (h region) are fenced from next gemm's h writes
        // by the post-argmax barrier; v is rewritten only after the next
        // post-gemm barrier, so the out-write read of v is safe.
    }
}

// ---------------------------------------------------------------------------
extern "C" void kernel_launch(void* const* d_in, const int* in_sizes, int n_in,
                              void* d_out, int out_size, void* d_ws, size_t ws_size,
                              hipStream_t stream)
{
    (void)in_sizes; (void)n_in; (void)out_size; (void)ws_size;
    const float* vertices   = (const float*)d_in[0];
    const int*   edge_index = (const int*)  d_in[1];
    const float* W1  = (const float*)d_in[2];
    const float* b1  = (const float*)d_in[3];
    const float* W2  = (const float*)d_in[4];
    const float* b2  = (const float*)d_in[5];
    const float* W3  = (const float*)d_in[6];
    const float* b3  = (const float*)d_in[7];
    const float* Wm1 = (const float*)d_in[8];
    const float* bm1 = (const float*)d_in[9];
    const float* Wm2 = (const float*)d_in[10];
    const float* bm2 = (const float*)d_in[11];

    char*  ws      = (char*)d_ws;
    int*   row_ptr = (int*)  ws;             // (NV+1)*4 = 2052 B
    int*   src_g   = (int*)  (ws + 4096);    // MT*4 = 34816 B
    float* dinv_g  = (float*)(ws + 40960);   // NV*4 = 2048 B

    hipLaunchKernelGGL(deepham_build_csr, dim3(1), dim3(1024), 0, stream,
                       edge_index, row_ptr, src_g, dinv_g);
    hipLaunchKernelGGL(deepham_main, dim3(1), dim3(1024), 0, stream,
                       vertices, W1, b1, W2, b2, W3, b3, Wm1, bm1, Wm2, bm2,
                       row_ptr, src_g, dinv_g, (float*)d_out);
}
